// MoELayer_50345606644208
// MI455X (gfx1250) — matrix operand order repaired and checked by reading, not compiled, not run on hardware
//
#include <hip/hip_runtime.h>
#include <math.h>

// ---------------------------------------------------------------------------
// MoE (top-2 of 8 experts) for MI455X / gfx1250.
// fp32 -> bf16 staging, v_wmma_f32_16x16x32_bf16 GEMMs, async global->LDS.
// TILE_M = 32: each B (weight) fragment feeds 2 WMMAs -> 2x arithmetic
// intensity vs the 16-row version (which was L2-bandwidth-bound).
// ---------------------------------------------------------------------------

typedef __bf16 bf16x16 __attribute__((ext_vector_type(16)));
typedef float  f32x8   __attribute__((ext_vector_type(8)));
typedef int    v4i     __attribute__((ext_vector_type(4)));

#define DEV static __device__ __forceinline__

constexpr int D_MODEL   = 512;
constexpr int D_FF      = 2048;
constexpr int N_EXP     = 8;
constexpr int N_TOK     = 4096;              // 4 * 1024
constexpr int TILE_M    = 32;                // tokens per block (2 WMMA rows)
constexpr int MAX_TILES = N_TOK / TILE_M;    // 128 (worst case one expert gets all)
constexpr int FF_CHUNK  = 256;               // ff processed per outer iteration
constexpr int XS        = D_MODEL + 8;       // padded LDS stride (bf16): 520*2B = 65*16B
constexpr int HS        = FF_CHUNK + 8;      // padded LDS stride: 264*2B = 33*16B

// ---- async global->LDS (gfx1250), guarded so we always compile ------------
#if defined(__has_builtin)
#  if __has_builtin(__builtin_amdgcn_global_load_async_to_lds_b128)
#    define HAVE_ASYNC_LDS 1
#  endif
#endif
#ifndef HAVE_ASYNC_LDS
#  define HAVE_ASYNC_LDS 0
#endif

#if defined(__has_builtin)
#  if __has_builtin(__builtin_amdgcn_s_wait_asynccnt)
#    define WAIT_ASYNC0() __builtin_amdgcn_s_wait_asynccnt(0)
#  endif
#endif
#ifndef WAIT_ASYNC0
#  define WAIT_ASYNC0() asm volatile("s_wait_asynccnt 0" ::: "memory")
#endif

#if HAVE_ASYNC_LDS
typedef __attribute__((address_space(1))) v4i* gv4i_p;   // global int4*
typedef __attribute__((address_space(3))) v4i* lv4i_p;   // LDS int4*
#endif

// ---- helpers ---------------------------------------------------------------
struct U32x8 { uint4 lo, hi; };

DEV bf16x16 make_frag(uint4 lo, uint4 hi) {
  U32x8 t{lo, hi};
  return __builtin_bit_cast(bf16x16, t);
}

DEV unsigned short f2bf(float f) {             // fp32 -> bf16, round-nearest-even
  unsigned u = __builtin_bit_cast(unsigned, f);
  u += 0x7FFFu + ((u >> 16) & 1u);
  return (unsigned short)(u >> 16);
}

DEV float gelu_exact(float v) { return 0.5f * v * (1.0f + erff(v * 0.70710678118654752f)); }

// A fragment (16x32 bf16, M x K) from LDS, matching the B fragment order:
// lane L (M = L&15): element j holds K = (L<16?0:16) + j
// => two contiguous 16B runs per lane.
DEV bf16x16 load_a_frag(const unsigned short* base, int stride, int kbase, int lane) {
  const int m  = lane & 15;
  const int lh = lane >> 4;
  const unsigned short* row = base + m * stride;
  uint4 lo = *(const uint4*)(row + kbase + lh * 16);
  uint4 hi = *(const uint4*)(row + kbase + lh * 16 + 8);
  return make_frag(lo, hi);
}

// B fragment (32x16 bf16, K x N) from global bf16 weights stored K-contiguous.
// lane L (N = L&15): element j holds K = (L<16?0:16) + j  => one 32B run.
DEV bf16x16 load_b_frag(const unsigned short* p) {
  const uint4* q = (const uint4*)p;
  return make_frag(q[0], q[1]);
}

DEV f32x8 wmma_bf16(bf16x16 a, bf16x16 b, f32x8 c) {
  return __builtin_amdgcn_wmma_f32_16x16x32_bf16(false, a, false, b, (short)0, c,
                                                 false, false);
}

// ---------------------------------------------------------------------------
// Kernel 1: convert x/w1/w2 to bf16 in workspace, zero output + expert counts.
// ---------------------------------------------------------------------------
__global__ __launch_bounds__(256) void moe_prep_kernel(
    const float* __restrict__ x, const float* __restrict__ w1,
    const float* __restrict__ w2,
    unsigned short* __restrict__ xbf, unsigned short* __restrict__ w1bf,
    unsigned short* __restrict__ w2bf, int* __restrict__ counts,
    float* __restrict__ out) {
  const size_t i  = (size_t)blockIdx.x * blockDim.x + threadIdx.x;
  constexpr size_t NW = (size_t)N_EXP * D_FF * D_MODEL;  // 8,388,608
  constexpr size_t NX = (size_t)N_TOK * D_MODEL;         // 2,097,152
  if (i < NW) { w1bf[i] = f2bf(w1[i]); w2bf[i] = f2bf(w2[i]); }
  if (i < NX) { xbf[i] = f2bf(x[i]); out[i] = 0.0f; }
  if (i < (size_t)N_EXP) counts[i] = 0;
}

// ---------------------------------------------------------------------------
// Kernel 2: router — logits, softmax, top-2, normalized gates, scatter lists.
// ---------------------------------------------------------------------------
__global__ __launch_bounds__(256) void moe_router_kernel(
    const float* __restrict__ x, const float* __restrict__ rw,
    int* __restrict__ counts, int* __restrict__ etok,
    float* __restrict__ egate) {
  const int t = blockIdx.x * 256 + threadIdx.x;
  if (t >= N_TOK) return;
  const float* xr = x + (size_t)t * D_MODEL;

  float lg[N_EXP];
#pragma unroll
  for (int e = 0; e < N_EXP; ++e) lg[e] = 0.0f;
  for (int c = 0; c < D_MODEL; ++c) {
    const float xv = xr[c];
#pragma unroll
    for (int e = 0; e < N_EXP; ++e) lg[e] = fmaf(xv, rw[e * D_MODEL + c], lg[e]);
  }
  float m = lg[0];
#pragma unroll
  for (int e = 1; e < N_EXP; ++e) m = fmaxf(m, lg[e]);
  float p[N_EXP], s = 0.0f;
#pragma unroll
  for (int e = 0; e < N_EXP; ++e) { p[e] = __expf(lg[e] - m); s += p[e]; }
  const float inv = 1.0f / s;
#pragma unroll
  for (int e = 0; e < N_EXP; ++e) p[e] *= inv;

  int i0 = 0; float p0 = p[0];
#pragma unroll
  for (int e = 1; e < N_EXP; ++e) if (p[e] > p0) { p0 = p[e]; i0 = e; }
  int i1 = -1; float p1 = -1.0f;
#pragma unroll
  for (int e = 0; e < N_EXP; ++e)
    if (e != i0 && p[e] > p1) { p1 = p[e]; i1 = e; }

  const float wsum = p0 + p1 + 1e-9f;
  const float g0 = p0 / wsum, g1 = p1 / wsum;

  int pos0 = atomicAdd(&counts[i0], 1);
  etok[i0 * N_TOK + pos0] = t;  egate[i0 * N_TOK + pos0] = g0;
  int pos1 = atomicAdd(&counts[i1], 1);
  etok[i1 * N_TOK + pos1] = t;  egate[i1 * N_TOK + pos1] = g1;
}

// ---------------------------------------------------------------------------
// Kernel 3: expert FFN.  One block = one 32-token tile of one expert.
// 256 threads = 8 waves.  Stage1: h = gelu(x*w1^T+b1) for a 256-wide ff chunk
// (each wave owns 32 ff cols x 32 tokens; B frag reused for 2 A frags).
// Stage2: acc += h*w2^T (each wave owns 64 of 512 output cols x 32 tokens).
// Epilogue: atomicAdd (acc+b2)*gate into out.
// ---------------------------------------------------------------------------
__global__ __launch_bounds__(256) void moe_ffn_kernel(
    const unsigned short* __restrict__ xbf,
    const unsigned short* __restrict__ w1bf,
    const unsigned short* __restrict__ w2bf,
    const float* __restrict__ b1, const float* __restrict__ b2,
    const int* __restrict__ counts, const int* __restrict__ etok,
    const float* __restrict__ egate, float* __restrict__ out) {
  const int e    = blockIdx.y;
  const int tile = blockIdx.x;
  const int cnt  = counts[e];
  if (tile * TILE_M >= cnt) return;

  __shared__ __align__(16) unsigned short x_sh[TILE_M * XS];   // 33,280 B
  __shared__ __align__(16) unsigned short h_sh[TILE_M * HS];   // 16,896 B
  __shared__ int   tok_sh[TILE_M];
  __shared__ float gate_sh[TILE_M];

  const int tid  = threadIdx.x;
  const int wave = tid >> 5;
  const int lane = tid & 31;
  const int lmod = lane & 15;
  const int lh   = lane >> 4;

  if (tid < TILE_M) {
    const int slot = tile * TILE_M + tid;
    if (slot < cnt) {
      tok_sh[tid]  = etok[e * N_TOK + slot];
      gate_sh[tid] = egate[e * N_TOK + slot];
    } else {
      tok_sh[tid]  = 0;       // valid row, gated to zero
      gate_sh[tid] = 0.0f;
    }
  }
  __syncthreads();

  // ---- stage the 32 token rows (bf16) into LDS: 128B per thread ----
  {
    const int r = tid >> 3, seg = tid & 7;                    // 32 rows x 8 segs
    const unsigned short* gsrc = xbf + (size_t)tok_sh[r] * D_MODEL + seg * 64;
    unsigned short* ldst = &x_sh[r * XS + seg * 64];
#if HAVE_ASYNC_LDS
#pragma unroll
    for (int i = 0; i < 8; ++i)
      __builtin_amdgcn_global_load_async_to_lds_b128(
          (gv4i_p)(gsrc + i * 8), (lv4i_p)(ldst + i * 8), 0, 0);
    WAIT_ASYNC0();
#else
#pragma unroll
    for (int i = 0; i < 8; ++i)
      *(uint4*)(ldst + i * 8) = *(const uint4*)(gsrc + i * 8);
#endif
  }
  __syncthreads();

  const unsigned short* w1e = w1bf + (size_t)e * D_FF * D_MODEL;
  const unsigned short* w2e = w2bf + (size_t)e * D_MODEL * D_FF;

  f32x8 acc[2][4] = {};   // [row-block ai][col tile t]: col = wave*64+t*16+lmod

  for (int ff0 = 0; ff0 < D_FF; ff0 += FF_CHUNK) {
    // ---------- stage 1: h[32 x 32] per wave, K = D_MODEL ----------
    f32x8 hacc[2][2] = {};
    const int ffA = ff0 + wave * 32;
    for (int k = 0; k < D_MODEL; k += 32) {
      const bf16x16 a0 = load_a_frag(x_sh, XS, k, lane);
      const bf16x16 a1 = load_a_frag(x_sh + 16 * XS, XS, k, lane);
#pragma unroll
      for (int t = 0; t < 2; ++t) {
        const unsigned short* bp =
            w1e + (size_t)(ffA + t * 16 + lmod) * D_MODEL + k + lh * 16;
        const bf16x16 b = load_b_frag(bp);        // reused for both row blocks
        hacc[0][t] = wmma_bf16(a0, b, hacc[0][t]);
        hacc[1][t] = wmma_bf16(a1, b, hacc[1][t]);
      }
    }
    // bias + exact GeLU -> bf16 h chunk in LDS
#pragma unroll
    for (int t = 0; t < 2; ++t) {
      const int ffc  = ffA + t * 16 + lmod;
      const float bv = b1[e * D_FF + ffc];
#pragma unroll
      for (int ai = 0; ai < 2; ++ai)
#pragma unroll
        for (int r = 0; r < 8; ++r) {
          const int row = ai * 16 + lh * 8 + r;   // C-layout row
          h_sh[row * HS + (ffc - ff0)] = f2bf(gelu_exact(hacc[ai][t][r] + bv));
        }
    }
    __syncthreads();

    // ---------- stage 2: acc += h * w2^T over this ff chunk ----------
    for (int kk = 0; kk < FF_CHUNK; kk += 32) {
      const bf16x16 a0 = load_a_frag(h_sh, HS, kk, lane);
      const bf16x16 a1 = load_a_frag(h_sh + 16 * HS, HS, kk, lane);
#pragma unroll
      for (int t = 0; t < 4; ++t) {
        const int col = wave * 64 + t * 16 + lmod;
        const unsigned short* bp =
            w2e + (size_t)col * D_FF + ff0 + kk + lh * 16;
        const bf16x16 b = load_b_frag(bp);        // reused for both row blocks
        acc[0][t] = wmma_bf16(a0, b, acc[0][t]);
        acc[1][t] = wmma_bf16(a1, b, acc[1][t]);
      }
    }
    __syncthreads();   // before next chunk overwrites h_sh
  }

  // ---------- epilogue: (acc + b2) * gate, scatter-add ----------
#pragma unroll
  for (int t = 0; t < 4; ++t) {
    const int col  = wave * 64 + t * 16 + lmod;
    const float bv = b2[e * D_MODEL + col];
#pragma unroll
    for (int ai = 0; ai < 2; ++ai)
#pragma unroll
      for (int r = 0; r < 8; ++r) {
        const int row = ai * 16 + lh * 8 + r;
        if (tile * TILE_M + row < cnt) {
          const float val = (acc[ai][t][r] + bv) * gate_sh[row];
          atomicAdd(&out[(size_t)tok_sh[row] * D_MODEL + col], val);
        }
      }
  }
}

// ---------------------------------------------------------------------------
// Host-side launcher.
// Inputs: 0=x [4,1024,512] f32, 1=router_w [8,512], 2=w1 [8,2048,512],
//         3=b1 [8,2048], 4=w2 [8,512,2048], 5=b2 [8,512].  Out: [4,1024,512].
// Workspace layout (~36.3 MB): x_bf16 | w1_bf16 | w2_bf16 | counts | tok | gate
// ---------------------------------------------------------------------------
extern "C" void kernel_launch(void* const* d_in, const int* in_sizes, int n_in,
                              void* d_out, int out_size, void* d_ws, size_t ws_size,
                              hipStream_t stream) {
  const float* x  = (const float*)d_in[0];
  const float* rw = (const float*)d_in[1];
  const float* w1 = (const float*)d_in[2];
  const float* b1 = (const float*)d_in[3];
  const float* w2 = (const float*)d_in[4];
  const float* b2 = (const float*)d_in[5];
  float* out = (float*)d_out;

  char* ws = (char*)d_ws;
  constexpr size_t NX = (size_t)N_TOK * D_MODEL;           // 2,097,152
  constexpr size_t NW = (size_t)N_EXP * D_FF * D_MODEL;    // 8,388,608
  unsigned short* xbf  = (unsigned short*)(ws);                          // 4 MiB
  unsigned short* w1bf = (unsigned short*)(ws + NX * 2);                 // 16 MiB
  unsigned short* w2bf = (unsigned short*)(ws + NX * 2 + NW * 2);        // 16 MiB
  char* tail = ws + NX * 2 + NW * 4;
  int*   counts = (int*)tail;                                            // 32 B -> pad 256
  int*   etok   = (int*)(tail + 256);                                    // 128 KiB
  float* egate  = (float*)(tail + 256 + (size_t)N_EXP * N_TOK * 4);      // 128 KiB

  // 1) convert + zero
  {
    const int nblk = (int)((NW + 255) / 256);   // 32768
    moe_prep_kernel<<<nblk, 256, 0, stream>>>(x, w1, w2, xbf, w1bf, w2bf,
                                              counts, out);
  }
  // 2) router / top-2 / scatter
  moe_router_kernel<<<N_TOK / 256, 256, 0, stream>>>(x, rw, counts, etok, egate);

  // 3) expert FFN (WMMA)
  {
    dim3 grid(MAX_TILES, N_EXP);
    moe_ffn_kernel<<<grid, 256, 0, stream>>>(xbf, w1bf, w2bf, b1, b2, counts,
                                             etok, egate, out);
  }
}
